// SpectralSpatialLinearAttention_39109972198166
// MI455X (gfx1250) — matrix operand order repaired and checked by reading, not compiled, not run on hardware
//
#include <hip/hip_runtime.h>
#include <hip/hip_bf16.h>
#include <math.h>

// ---------------- problem constants ----------------
#define BATCH 8
#define SEQ   16384
#define CH    256
#define HEADS 8
#define HDIM  32
#define GHID  64
#define MROWS (BATCH * SEQ)   // 131072

typedef __bf16 bf16;
typedef __attribute__((ext_vector_type(16))) __bf16 bf16x16;
typedef __attribute__((ext_vector_type(8)))  float  v8f;

union FragBF { bf16x16 v; uint4 q[2]; };

// ---- CDNA5 async global->LDS path (guarded; falls back to plain LDS copy) ----
#if defined(__has_builtin)
#if __has_builtin(__builtin_amdgcn_global_load_async_to_lds_b128) && \
    __has_builtin(__builtin_amdgcn_s_wait_asynccnt)
#define HAVE_ASYNC_LDS 1
#endif
#endif

// Exact pointee type from the hipcc diagnostic: vector_size(16) int, AS1 / AS3.
typedef int v4i_vs __attribute__((vector_size(16)));
typedef __attribute__((address_space(1))) v4i_vs* as1_v4i;
typedef __attribute__((address_space(3))) v4i_vs* as3_v4i;

__device__ __forceinline__ bf16 f2bf(float f) {
  unsigned u = __float_as_uint(f);
  u += 0x7fffu + ((u >> 16) & 1u);            // round-to-nearest-even
  unsigned short h = (unsigned short)(u >> 16);
  union { unsigned short s; bf16 b; } cv; cv.s = h;
  return cv.b;
}

// Stage NELEM bf16 (contiguous in global) into LDS with the whole block.
template <int NELEM>
__device__ __forceinline__ void stage_panel(const bf16* __restrict__ g,
                                            bf16* __restrict__ s) {
#if defined(HAVE_ASYNC_LDS)
  for (int i = threadIdx.x; i < NELEM / 8; i += 128) {
    __builtin_amdgcn_global_load_async_to_lds_b128(
        (as1_v4i)(g + (size_t)i * 8),
        (as3_v4i)(s + (size_t)i * 8), 0, 0);
  }
  __builtin_amdgcn_s_wait_asynccnt(0);
  __syncthreads();
#else
  for (int i = threadIdx.x; i < NELEM / 8; i += 128)
    *(uint4*)(s + (size_t)i * 8) = *(const uint4*)(g + (size_t)i * 8);
  __syncthreads();
#endif
}

// ---------------------------------------------------------------------------
// Wave computes a 16x64 strip of C: A fragment loaded once per k-step (global,
// row-major bf16, lda == KDIM), 4 B fragments read from the LDS-resident
// Bt panel (row-major [64][KDIM] bf16), 4 WMMA per k-step.
// A frag (16x32): lanes 0-15 hold K=0..15 of row M=lane, lanes 16-31 K=16..31.
// B frag (32x16): lanes 0-15 hold K=0..15 of col N=lane, lanes 16-31 K=16..31.
// ---------------------------------------------------------------------------
template <int KDIM>
__device__ __forceinline__ void block_gemm(const bf16* __restrict__ A, long m0w,
                                           const bf16* __restrict__ sB,
                                           v8f acc[4]) {
  int lane = threadIdx.x & 31;
  int la   = lane & 15;
  int kbA  = (lane >> 4) << 3;   // 0 or 8
  int kbB  = (lane >> 4) << 4;   // 0 or 16
  const bf16* ap = A + (size_t)(m0w + la) * KDIM;
#pragma unroll
  for (int k0 = 0; k0 < KDIM; k0 += 32) {
    FragBF a;
    a.q[0] = *(const uint4*)(ap + k0 + kbB);
    a.q[1] = *(const uint4*)(ap + k0 + kbB + 8);
    if (k0 + 32 < KDIM) __builtin_prefetch(ap + k0 + 64, 0, 1);
#pragma unroll
    for (int nn = 0; nn < 4; ++nn) {
      FragBF b;
      const bf16* bp = sB + (size_t)(nn * 16 + la) * KDIM + k0 + kbB;
      b.q[0] = *(const uint4*)(bp);
      b.q[1] = *(const uint4*)(bp + 8);
      acc[nn] = __builtin_amdgcn_wmma_f32_16x16x32_bf16(
          false, a.v, false, b.v, (short)0, acc[nn], false, false);
    }
  }
}

// ---------------- weight transpose + f32 -> bf16 ----------------
__global__ void k_transpose(const float* __restrict__ in, bf16* __restrict__ out,
                            int R, int Cc) {
  int i = blockIdx.x * blockDim.x + threadIdx.x;
  if (i >= R * Cc) return;
  int r = i / Cc, c = i % Cc;
  out[(size_t)c * R + r] = f2bf(in[i]);
}

// ---------------- LayerNorm (wave per row) + x cast ----------------
__global__ void k_ln(const float* __restrict__ x, const float* __restrict__ g,
                     const float* __restrict__ bta, bf16* __restrict__ xb,
                     bf16* __restrict__ xn) {
  int wave = (int)((blockIdx.x * blockDim.x + threadIdx.x) >> 5);
  int lane = threadIdx.x & 31;
  if (wave >= MROWS) return;
  const float* xr = x + (size_t)wave * CH + lane * 8;
  float4 a4 = *(const float4*)xr;
  float4 b4 = *(const float4*)(xr + 4);
  float v[8] = {a4.x, a4.y, a4.z, a4.w, b4.x, b4.y, b4.z, b4.w};
  float s = 0.f, s2 = 0.f;
  for (int i = 0; i < 8; ++i) { s += v[i]; s2 += v[i] * v[i]; }
  for (int m = 16; m >= 1; m >>= 1) {
    s  += __shfl_xor(s,  m, 32);
    s2 += __shfl_xor(s2, m, 32);
  }
  float mu  = s * (1.f / CH);
  float var = s2 * (1.f / CH) - mu * mu;
  float inv = rsqrtf(var + 1e-5f);
  union { bf16 h[8]; uint4 q; } ob, on;
  for (int i = 0; i < 8; ++i) {
    int c = lane * 8 + i;
    ob.h[i] = f2bf(v[i]);
    on.h[i] = f2bf((v[i] - mu) * inv * g[c] + bta[c]);
  }
  *(uint4*)(xb + (size_t)wave * CH + lane * 8) = ob.q;
  *(uint4*)(xn + (size_t)wave * CH + lane * 8) = on.q;
}

// ---------------- qkv GEMM: [M,256] x [256,768]; split epilogue ----------------
__global__ void __launch_bounds__(128)
k_qkv(const bf16* __restrict__ xb, const bf16* __restrict__ WqkvT,
      bf16* __restrict__ q, float* __restrict__ kk, bf16* __restrict__ vv) {
  __shared__ bf16 sB[64 * 256];
  int nb = blockIdx.x % 12, mb = blockIdx.x / 12;
  long m0 = (long)mb * 64;
  int  n0 = nb * 64;
  stage_panel<64 * 256>(WqkvT + (size_t)n0 * 256, sB);
  v8f acc[4] = {};
  int w = threadIdx.x >> 5, lane = threadIdx.x & 31;
  long m0w = m0 + w * 16;
  block_gemm<256>(xb, m0w, sB, acc);
  long row = m0w + ((lane >> 4) << 3);
  int  la  = lane & 15;
#pragma unroll
  for (int nn = 0; nn < 4; ++nn) {
    int col = n0 + nn * 16 + la;
    if (n0 < 256) {
      for (int r = 0; r < 8; ++r)
        q[(size_t)(row + r) * CH + col] = f2bf(acc[nn][r]);
    } else if (n0 < 512) {
      int c2 = col - 256;
      for (int r = 0; r < 8; ++r)
        kk[(size_t)(row + r) * CH + c2] = acc[nn][r];
    } else {
      int c2 = col - 512;
      for (int r = 0; r < 8; ++r)
        vv[(size_t)(row + r) * CH + c2] = f2bf(acc[nn][r]);
    }
  }
}

// ---------------- softmax stats over N per (b, h*32+d) column ----------------
__global__ void k_red(const float* __restrict__ kk, float* __restrict__ kmax,
                      float* __restrict__ ksum) {
  __shared__ float sm[256];
  int b  = blockIdx.x >> 8;
  int hd = blockIdx.x & 255;
  const float* base = kk + (size_t)b * SEQ * CH + hd;
  float m = -1e30f;
  for (int n = threadIdx.x; n < SEQ; n += 256) m = fmaxf(m, base[(size_t)n * CH]);
  sm[threadIdx.x] = m; __syncthreads();
  for (int s = 128; s >= 1; s >>= 1) {
    if ((int)threadIdx.x < s) sm[threadIdx.x] = fmaxf(sm[threadIdx.x], sm[threadIdx.x + s]);
    __syncthreads();
  }
  float mx = sm[0]; __syncthreads();
  float su = 0.f;
  for (int n = threadIdx.x; n < SEQ; n += 256) su += expf(base[(size_t)n * CH] - mx);
  sm[threadIdx.x] = su; __syncthreads();
  for (int s = 128; s >= 1; s >>= 1) {
    if ((int)threadIdx.x < s) sm[threadIdx.x] += sm[threadIdx.x + s];
    __syncthreads();
  }
  if (threadIdx.x == 0) { kmax[blockIdx.x] = mx; ksum[blockIdx.x] = sm[0]; }
}

// ---------------- context = softmax(k)^T v, 32x32 per (b,h), WMMA over N ----------------
__global__ void k_context(const float* __restrict__ kk, const bf16* __restrict__ vv,
                          const float* __restrict__ kmax, const float* __restrict__ ksum,
                          float* __restrict__ ctx) {
  int bh = blockIdx.x;
  int b = bh >> 3, h = bh & 7;
  int w = threadIdx.x >> 5, lane = threadIdx.x & 31;
  int ti = (w & 1) * 16, tj = (w >> 1) * 16;        // quadrant of 32x32 output
  int la  = lane & 15;
  int kbA = (lane >> 4) << 3;
  int kbB = (lane >> 4) << 4;
  int d    = ti + la;                               // A row  (k^T row = d)
  int colv = tj + la;                               // B col  (v dim)
  float rm = kmax[b * 256 + h * HDIM + d];
  float rs = 1.f / ksum[b * 256 + h * HDIM + d];
  const float* kbase = kk + (size_t)b * SEQ * CH + h * HDIM + d;
  const bf16*  vbase = vv + (size_t)b * SEQ * CH + h * HDIM + colv;
  v8f acc = {};
  for (int n0 = 0; n0 < SEQ; n0 += 32) {
    FragBF a, bb;
    for (int j = 0; j < 8; ++j) {
      a.v[j]     = f2bf(expf(kbase[(size_t)(n0 + kbB + j)      * CH] - rm) * rs);
      a.v[8 + j] = f2bf(expf(kbase[(size_t)(n0 + kbB + 8 + j) * CH] - rm) * rs);
    }
    for (int j = 0; j < 16; ++j)
      bb.v[j] = vbase[(size_t)(n0 + kbB + j) * CH];
    acc = __builtin_amdgcn_wmma_f32_16x16x32_bf16(false, a.v, false, bb.v,
                                                  (short)0, acc, false, false);
  }
  int rowb = ti + ((lane >> 4) << 3);
  for (int r = 0; r < 8; ++r)
    ctx[(size_t)bh * 1024 + (size_t)(rowb + r) * 32 + tj + la] = acc[r];
}

// ---------------- out = q @ context per (b,h) ----------------
__global__ void k_out(const bf16* __restrict__ q, const float* __restrict__ ctx,
                      float* __restrict__ attn) {
  int t = (int)((blockIdx.x * blockDim.x + threadIdx.x) >> 5);
  int lane = threadIdx.x & 31;
  int vt = t & 1;
  int nt = (t >> 1) & 1023;
  int bh = t >> 11;
  int b = bh >> 3, h = bh & 7;
  long m0 = (long)b * SEQ + nt * 16;
  int  v0 = vt * 16;
  int la  = lane & 15;
  int kbA = (lane >> 4) << 3;
  int kbB = (lane >> 4) << 4;
  FragBF a, bb;
  const bf16* ap = q + (size_t)(m0 + la) * CH + h * HDIM;
  a.q[0] = *(const uint4*)(ap + kbB);
  a.q[1] = *(const uint4*)(ap + kbB + 8);
  for (int j = 0; j < 16; ++j)
    bb.v[j] = f2bf(ctx[(size_t)bh * 1024 + (size_t)(kbB + j) * 32 + v0 + la]);
  v8f acc = {};
  acc = __builtin_amdgcn_wmma_f32_16x16x32_bf16(false, a.v, false, bb.v,
                                                (short)0, acc, false, false);
  long row = m0 + ((lane >> 4) << 3);
  int  col = h * HDIM + v0 + la;
  for (int r = 0; r < 8; ++r) attn[(size_t)(row + r) * CH + col] = acc[r];
}

// ---------------- gate1: gelu(xn @ Wg1 + bg1), N = 64 (one n-block) ----------------
__global__ void __launch_bounds__(128)
k_gate1(const bf16* __restrict__ xn, const bf16* __restrict__ Wg1T,
        const float* __restrict__ bg1, bf16* __restrict__ hg) {
  __shared__ bf16 sB[64 * 256];
  long m0 = (long)blockIdx.x * 64;
  stage_panel<64 * 256>(Wg1T, sB);
  v8f acc[4] = {};
  int w = threadIdx.x >> 5, lane = threadIdx.x & 31;
  long m0w = m0 + w * 16;
  block_gemm<256>(xn, m0w, sB, acc);
  long row = m0w + ((lane >> 4) << 3);
  int  la  = lane & 15;
#pragma unroll
  for (int nn = 0; nn < 4; ++nn) {
    int col = nn * 16 + la;
    float bb = bg1[col];
    for (int r = 0; r < 8; ++r) {
      float vx = acc[nn][r] + bb;
      float ge = 0.5f * vx * (1.f + erff(vx * 0.70710678118f));
      hg[(size_t)(row + r) * GHID + col] = f2bf(ge);
    }
  }
}

// ---------------- gate2: sigmoid(hg @ Wg2 + bg2) * attn ----------------
__global__ void __launch_bounds__(128)
k_gate2(const bf16* __restrict__ hg, const bf16* __restrict__ Wg2T,
        const float* __restrict__ bg2, const float* __restrict__ attn,
        bf16* __restrict__ y) {
  __shared__ bf16 sB[64 * 64];
  int nb = blockIdx.x & 3, mb = blockIdx.x >> 2;
  long m0 = (long)mb * 64;
  int  n0 = nb * 64;
  stage_panel<64 * 64>(Wg2T + (size_t)n0 * GHID, sB);
  v8f acc[4] = {};
  int w = threadIdx.x >> 5, lane = threadIdx.x & 31;
  long m0w = m0 + w * 16;
  block_gemm<64>(hg, m0w, sB, acc);
  long row = m0w + ((lane >> 4) << 3);
  int  la  = lane & 15;
#pragma unroll
  for (int nn = 0; nn < 4; ++nn) {
    int col = n0 + nn * 16 + la;
    float bb = bg2[col];
    for (int r = 0; r < 8; ++r) {
      float gate = 1.f / (1.f + expf(-(acc[nn][r] + bb)));
      y[(size_t)(row + r) * CH + col] = f2bf(gate * attn[(size_t)(row + r) * CH + col]);
    }
  }
}

// ---------------- proj: y @ Wproj + bproj -> d_out (f32) ----------------
__global__ void __launch_bounds__(128)
k_proj(const bf16* __restrict__ y, const bf16* __restrict__ WprojT,
       const float* __restrict__ bproj, float* __restrict__ out) {
  __shared__ bf16 sB[64 * 256];
  int nb = blockIdx.x & 3, mb = blockIdx.x >> 2;
  long m0 = (long)mb * 64;
  int  n0 = nb * 64;
  stage_panel<64 * 256>(WprojT + (size_t)n0 * CH, sB);
  v8f acc[4] = {};
  int w = threadIdx.x >> 5, lane = threadIdx.x & 31;
  long m0w = m0 + w * 16;
  block_gemm<256>(y, m0w, sB, acc);
  long row = m0w + ((lane >> 4) << 3);
  int  la  = lane & 15;
#pragma unroll
  for (int nn = 0; nn < 4; ++nn) {
    int col = n0 + nn * 16 + la;
    float bb = bproj[col];
    for (int r = 0; r < 8; ++r)
      out[(size_t)(row + r) * CH + col] = acc[nn][r] + bb;
  }
}

// ---------------------------------------------------------------------------
extern "C" void kernel_launch(void* const* d_in, const int* in_sizes, int n_in,
                              void* d_out, int out_size, void* d_ws, size_t ws_size,
                              hipStream_t stream) {
  const float* x     = (const float*)d_in[0];
  const float* Wqkv  = (const float*)d_in[1];
  const float* Wproj = (const float*)d_in[2];
  const float* bproj = (const float*)d_in[3];
  const float* ln_g  = (const float*)d_in[4];
  const float* ln_b  = (const float*)d_in[5];
  const float* Wg1   = (const float*)d_in[6];
  const float* bg1   = (const float*)d_in[7];
  const float* Wg2   = (const float*)d_in[8];
  const float* bg2   = (const float*)d_in[9];
  float* out = (float*)d_out;

  char* p = (char*)d_ws;
  auto alloc = [&](size_t bytes) -> char* {
    char* r = p;
    p += (bytes + 255) & ~(size_t)255;
    return r;
  };
  bf16*  xb     = (bf16*) alloc((size_t)MROWS * CH * 2);
  bf16*  xn     = (bf16*) alloc((size_t)MROWS * CH * 2);
  bf16*  qb     = (bf16*) alloc((size_t)MROWS * CH * 2);
  float* kk     = (float*)alloc((size_t)MROWS * CH * 4);
  bf16*  vb     = (bf16*) alloc((size_t)MROWS * CH * 2);
  float* kmax   = (float*)alloc(2048 * 4);
  float* ksum   = (float*)alloc(2048 * 4);
  float* ctx    = (float*)alloc(64 * 1024 * 4);
  float* attn   = (float*)alloc((size_t)MROWS * CH * 4);
  bf16*  hg     = (bf16*) alloc((size_t)MROWS * GHID * 2);
  bf16*  yb     = (bf16*) alloc((size_t)MROWS * CH * 2);
  bf16*  WqkvT  = (bf16*) alloc((size_t)768 * 256 * 2);
  bf16*  Wg1T   = (bf16*) alloc((size_t)64 * 256 * 2);
  bf16*  Wg2T   = (bf16*) alloc((size_t)256 * 64 * 2);
  bf16*  WprojT = (bf16*) alloc((size_t)256 * 256 * 2);

  // weight prep
  k_transpose<<<(256 * 768 + 255) / 256, 256, 0, stream>>>(Wqkv,  WqkvT,  256, 768);
  k_transpose<<<(256 * 64  + 255) / 256, 256, 0, stream>>>(Wg1,   Wg1T,   256, 64);
  k_transpose<<<(64  * 256 + 255) / 256, 256, 0, stream>>>(Wg2,   Wg2T,   64,  256);
  k_transpose<<<(256 * 256 + 255) / 256, 256, 0, stream>>>(Wproj, WprojT, 256, 256);

  // LN + bf16 cast of x
  k_ln<<<MROWS / 8, 256, 0, stream>>>(x, ln_g, ln_b, xb, xn);

  // attention path
  k_qkv<<<2048 * 12, 128, 0, stream>>>(xb, WqkvT, qb, kk, vb);
  k_red<<<2048, 256, 0, stream>>>(kk, kmax, ksum);
  k_context<<<64, 128, 0, stream>>>(kk, vb, kmax, ksum, ctx);
  k_out<<<131072 / 4, 128, 0, stream>>>(qb, ctx, attn);

  // gate path
  k_gate1<<<2048, 128, 0, stream>>>(xn, Wg1T, bg1, hg);
  k_gate2<<<2048 * 4, 128, 0, stream>>>(hg, Wg2T, bg2, attn, yb);

  // final projection
  k_proj<<<2048 * 4, 128, 0, stream>>>(yb, WprojT, bproj, out);
}
